// HierarchicalAttention_85581518340360
// MI455X (gfx1250) — matrix operand order repaired and checked by reading, not compiled, not run on hardware
//
#include <hip/hip_runtime.h>
#include <hip/hip_bf16.h>

// ---------------------------------------------------------------------------
// HierarchicalAttention for MI455X (gfx1250), wave32 + v_wmma_f32_16x16x32_bf16
// Fully fused: per 16-row tile of x, per head:
//   q = x@Wq_h (WMMA K=640) -> hie softmax(2) -> s = q@k^T (WMMA K=64)
//   -> softmax(256) -> o = attn@v (WMMA K=256) -> combine -> merged-head tile
//   -> out = oc@Wo + bo (WMMA K=512)
// B operands pre-swizzled to WMMA fragment layout (32B/lane b128 loads),
// A fragments staged in LDS with vectorized b128 packing, 4 accumulators
// per A-fragment for WMMA ILP. Native bf16 converts (v_cvt*_bf16_f32).
// ---------------------------------------------------------------------------

typedef __attribute__((ext_vector_type(16))) __bf16    v16bf;
typedef __attribute__((ext_vector_type(2)))  __bf16    v2bf;
typedef __attribute__((ext_vector_type(8)))  float     v8f;
typedef __attribute__((ext_vector_type(4)))  unsigned  u32x4;
typedef __attribute__((ext_vector_type(4)))  float     f32x4;

union AF {                 // one WMMA 16-bit operand fragment (32 B per lane)
  v16bf          v;
  unsigned short u[16];
  u32x4          q[2];
};

// float -> bf16 via native convert (1 VALU op, RNE)
__device__ __forceinline__ unsigned short f2bf(float f) {
  union { __bf16 b; unsigned short s; } c;
  c.b = (__bf16)f;
  return c.s;
}
// 2x float -> packed bf16 dword (lets backend use v_cvt_pk_bf16_f32)
__device__ __forceinline__ unsigned pk2(float a, float b) {
  union { v2bf v; unsigned u; } c;
  c.v[0] = (__bf16)a;
  c.v[1] = (__bf16)b;
  return c.u;
}

// 16-bit A-matrix 16x32 (MxK) VGPR layout, flattened lane*16+elem
__device__ __forceinline__ int idxA(int m, int kk) {
  int lane = m + (((kk >> 3) & 1) << 4);
  int elem = (kk & 7) + (((kk >> 4) & 1) << 3);
  return lane * 16 + elem;
}
// 16-bit B-matrix 32x16 (KxN) VGPR layout (same K order as idxA)
__device__ __forceinline__ int idxB(int nn, int kk) {
  return (nn + (((kk >> 3) & 1) << 4)) * 16 + (kk & 7) + (((kk >> 4) & 1) << 3);
}

__device__ __forceinline__ void loadFrag(AF& f, const unsigned short* p, int lane) {
  const u32x4* v = (const u32x4*)p;
  f.q[0] = v[lane * 2]; f.q[1] = v[lane * 2 + 1];
}

// ---------------------------------------------------------------------------
__global__ void zero_f32(float* p, int n) {
  int i = blockIdx.x * blockDim.x + threadIdx.x;
  if (i < n) p[i] = 0.f;
}

__global__ void pool_sum(const float* __restrict__ x, float* __restrict__ pooled) {
  int b = blockIdx.y;
  int col = threadIdx.x;                 // 640 threads
  int r0 = blockIdx.x * 128;             // 64 blocks x 128 rows
  const float* base = x + ((size_t)b * 8192 + r0) * 640 + col;
  float s = 0.f;
  for (int r = 0; r < 128; ++r) s += base[(size_t)r * 640];
  atomicAdd(&pooled[b * 640 + col], s);
}

// KxNtot f32 weight -> per-head B-fragment bf16 layout out[head][K/32][headN/16][512]
__global__ void conv_w(const float* __restrict__ W, int K, int Ntot, int headN,
                       unsigned short* __restrict__ out) {
  int idx = blockIdx.x * 256 + threadIdx.x;
  if (idx >= K * Ntot) return;
  int perHead = K * headN;
  int head = idx / perHead;
  int rem  = idx % perHead;
  int kkg  = rem / headN;
  int c    = rem % headN;
  int ntiles = headN >> 4;
  size_t frag = (size_t)((head * (K >> 5) + (kkg >> 5)) * ntiles + (c >> 4));
  out[frag * 512 + idxB(c & 15, kkg & 31)] =
      f2bf(W[(size_t)kkg * Ntot + head * headN + c]);
}

// k/v projections stored directly as WMMA B-fragments (bf16)
__global__ void proj_kv(const float* __restrict__ ctx,
                        const float* __restrict__ Wk,
                        const float* __restrict__ Wv,
                        unsigned short* __restrict__ kF,
                        unsigned short* __restrict__ vF) {
  int idx = blockIdx.x * 256 + threadIdx.x;   // 2^19 total
  int d   = idx & 63;
  int key = (idx >> 6) & 255;
  int h   = (idx >> 14) & 7;
  int b   = idx >> 17;
  int col = h * 64 + d;
  const float* crow = ctx + ((size_t)b * 256 + key) * 640;
  float ka = 0.f, va = 0.f;
  for (int c = 0; c < 640; ++c) {
    float cv = crow[c];
    ka += cv * Wk[(size_t)c * 512 + col];
    va += cv * Wv[(size_t)c * 512 + col];
  }
  kF[(size_t)(((b * 8 + h) * 2 + (d >> 5)) * 16 + (key >> 4)) * 512 +
     idxB(key & 15, d & 31)] = f2bf(ka);
  vF[(size_t)(((b * 8 + h) * 8 + (key >> 5)) * 4 + (d >> 4)) * 512 +
     idxB(d & 15, key & 31)] = f2bf(va);
}

__global__ void hie_proj(const float* __restrict__ ctx,
                         const float* __restrict__ pooled,
                         const float* __restrict__ Wk,
                         const float* __restrict__ Wk_,
                         const float* __restrict__ Wv_,
                         float* __restrict__ hieK, float* __restrict__ bgV) {
  int idx = blockIdx.x * 256 + threadIdx.x;
  if (idx >= 2048) return;
  int d = idx & 63, h = (idx >> 6) & 7, b = idx >> 9;
  int col = h * 64 + d;
  const float* crow = ctx + (size_t)b * 256 * 640;   // context[b][0][:]
  const float* prow = pooled + b * 640;
  float fg = 0.f, bgk = 0.f, bgv = 0.f;
  for (int c = 0; c < 640; ++c) {
    float pv = prow[c] * (1.f / 8192.f);             // mean
    fg  += crow[c] * Wk [(size_t)c * 512 + col];
    bgk += pv      * Wk_[(size_t)c * 512 + col];
    bgv += pv      * Wv_[(size_t)c * 512 + col];
  }
  hieK[((b * 8 + h) * 2    ) * 64 + d] = fg;
  hieK[((b * 8 + h) * 2 + 1) * 64 + d] = bgk;
  bgV [( b * 8 + h         ) * 64 + d] = bgv;
}

// ---------------------------------------------------------------------------
// Dynamic LDS layout (bytes):
//   xf  @      0 : 2 waves * 20 frag * 512 ush = 40960
//   scr @  40960 : 2 * 16*256 f32              = 32768
//   af  @  73728 : 2 * 8 frag * 512 ush        = 16384
//   oc  @  90112 : 2 * 16 frag * 512 ush       = 32768
//   hw  @ 122880 : 2 * 16*2 f32                =   256
#define SMEM_BYTES 123136

__global__ void __launch_bounds__(64, 1)
main_attn(const float* __restrict__ x,
          const unsigned short* __restrict__ WqF,
          const unsigned short* __restrict__ kF,
          const unsigned short* __restrict__ vF,
          const float* __restrict__ hieK,
          const float* __restrict__ bgV,
          const unsigned short* __restrict__ WoF,
          const float* __restrict__ bo,
          float* __restrict__ out) {
  extern __shared__ char smem[];
  unsigned short* xf  = (unsigned short*)(smem);
  float*          scr = (float*)         (smem + 40960);
  unsigned short* af  = (unsigned short*)(smem + 73728);
  unsigned short* oc  = (unsigned short*)(smem + 90112);
  float*          hwp = (float*)         (smem + 122880);

  const int b    = blockIdx.y;
  const int w    = threadIdx.x >> 5;
  const int lane = threadIdx.x & 31;
  const int row0 = blockIdx.x * 32 + w * 16;
  const int hi   = (lane >> 4) & 1;
  const int nn   = lane & 15;

  float*          scrW = scr + w * 4096;   // [16][256] f32
  unsigned short* afW  = af  + w * 4096;   // 8 A-frags
  unsigned short* ocW  = oc  + w * 8192;   // 16 A-frags
  float*          hwW  = hwp + w * 32;     // [16][2] f32

  // ---- stage x tile -> bf16 A-fragments in LDS (vectorized, 2xb128/lane) ----
  {
    const float* xrow = x + ((size_t)b * 8192 + row0) * 640;
    int m  = lane >> 1;
    int c0 = (lane & 1) * 16;      // 16 contiguous K values per lane
    int eo = c0 >> 1;              // elem offset (0 or 8) within lane row
    for (int kt = 0; kt < 20; ++kt) {
      const f32x4* s4 = (const f32x4*)(xrow + (size_t)m * 640 + kt * 32 + c0);
      f32x4 r0 = s4[0], r1 = s4[1], r2 = s4[2], r3 = s4[3];
      u32x4 pA, pB;
      pA[0] = pk2(r0[0], r0[1]); pA[1] = pk2(r0[2], r0[3]);
      pA[2] = pk2(r1[0], r1[1]); pA[3] = pk2(r1[2], r1[3]);
      pB[0] = pk2(r2[0], r2[1]); pB[1] = pk2(r2[2], r2[3]);
      pB[2] = pk2(r3[0], r3[1]); pB[3] = pk2(r3[2], r3[3]);
      unsigned short* dst = xf + (w * 20 + kt) * 512;
      *(u32x4*)(dst + m * 16 + eo)        = pA;   // kk c0..c0+7
      *(u32x4*)(dst + (m + 16) * 16 + eo) = pB;   // kk c0+8..c0+15
    }
  }
  __syncthreads();

  for (int h = 0; h < 8; ++h) {
    // ---- q = x_tile @ Wq_h : 4 independent accumulators per A-frag ----
    {
      v8f a0 = {}, a1 = {}, a2 = {}, a3 = {};
      for (int kt = 0; kt < 20; ++kt) {
        AF a, b0, b1, b2, b3;
        loadFrag(a, xf + (w * 20 + kt) * 512, lane);
        const unsigned short* wb = WqF + (size_t)(h * 20 + kt) * 4 * 512;
        loadFrag(b0, wb,          lane);
        loadFrag(b1, wb + 512,    lane);
        loadFrag(b2, wb + 1024,   lane);
        loadFrag(b3, wb + 1536,   lane);
        a0 = __builtin_amdgcn_wmma_f32_16x16x32_bf16(false, a.v, false, b0.v, (short)0, a0, false, false);
        a1 = __builtin_amdgcn_wmma_f32_16x16x32_bf16(false, a.v, false, b1.v, (short)0, a1, false, false);
        a2 = __builtin_amdgcn_wmma_f32_16x16x32_bf16(false, a.v, false, b2.v, (short)0, a2, false, false);
        a3 = __builtin_amdgcn_wmma_f32_16x16x32_bf16(false, a.v, false, b3.v, (short)0, a3, false, false);
      }
      // stage scaled q to scrW[16][64] (C-frag layout scatter)
#pragma unroll
      for (int r = 0; r < 8; ++r) {
        int mm = (r + hi * 8) * 64 + nn;
        scrW[mm]      = a0[r] * 0.125f;
        scrW[mm + 16] = a1[r] * 0.125f;
        scrW[mm + 32] = a2[r] * 0.125f;
        scrW[mm + 48] = a3[r] * 0.125f;
      }
    }
    __syncthreads();

    // ---- hierarchical softmax over the 2 keys (one lane per row) ----
    if (lane < 16) {
      const float* hk = hieK + (size_t)((b * 8 + h) * 2) * 64;
      float l0 = 0.f, l1 = 0.f;
      for (int dd = 0; dd < 64; ++dd) {
        float qv = scrW[lane * 64 + dd];
        l0 += qv * hk[dd];
        l1 += qv * hk[64 + dd];
      }
      float mx = fmaxf(l0, l1);
      float e0 = __expf(l0 - mx), e1 = __expf(l1 - mx);
      float inv = 1.f / (e0 + e1);
      hwW[lane * 2]     = e0 * inv;
      hwW[lane * 2 + 1] = e1 * inv;
    }

    // ---- repack q (f32 LDS) into bf16 A-fragments (vectorized) ----
    AF qa[2];
    {
      const f32x4* qrow = (const f32x4*)(scrW + (lane & 15) * 64);
#pragma unroll
      for (int kt = 0; kt < 2; ++kt) {
        int base = kt * 8 + hi * 2;                // f32x4 units
        f32x4 l0 = qrow[base], l1 = qrow[base + 1];        // kk = kt*32+hi*8 ..+8
        f32x4 h0 = qrow[base + 4], h1 = qrow[base + 5];    // kk = +16
        qa[kt].q[0][0] = pk2(l0[0], l0[1]); qa[kt].q[0][1] = pk2(l0[2], l0[3]);
        qa[kt].q[0][2] = pk2(l1[0], l1[1]); qa[kt].q[0][3] = pk2(l1[2], l1[3]);
        qa[kt].q[1][0] = pk2(h0[0], h0[1]); qa[kt].q[1][1] = pk2(h0[2], h0[3]);
        qa[kt].q[1][2] = pk2(h1[0], h1[1]); qa[kt].q[1][3] = pk2(h1[2], h1[3]);
      }
    }
    __syncthreads();

    // ---- scores = q @ k^T : 16 key-tiles, K=64 ----
    const unsigned short* kFh = kF + (size_t)((b * 8 + h) * 2 * 16) * 512;
    for (int nt = 0; nt < 16; ++nt) {
      AF c0, c1;
      loadFrag(c0, kFh + (size_t)nt * 512,        lane);
      loadFrag(c1, kFh + (size_t)(16 + nt) * 512, lane);
      v8f s = {};
      s = __builtin_amdgcn_wmma_f32_16x16x32_bf16(false, qa[0].v, false, c0.v, (short)0, s, false, false);
      s = __builtin_amdgcn_wmma_f32_16x16x32_bf16(false, qa[1].v, false, c1.v, (short)0, s, false, false);
      int n = nt * 16 + nn;
#pragma unroll
      for (int r = 0; r < 8; ++r) scrW[(r + hi * 8) * 256 + n] = s[r];
    }
    __syncthreads();

    // ---- softmax over 256 keys: 2 lanes per row, vectorized LDS ----
    {
      int row = lane & 15;
      float* srow = scrW + row * 256 + hi * 128;
      const f32x4* sv = (const f32x4*)srow;
      float mx = -3.0e38f;
#pragma unroll
      for (int t = 0; t < 32; ++t) {
        f32x4 vv = sv[t];
        mx = fmaxf(mx, fmaxf(fmaxf(vv[0], vv[1]), fmaxf(vv[2], vv[3])));
      }
      mx = fmaxf(mx, __shfl_xor(mx, 16, 32));
      float sum = 0.f;
      f32x4* svw = (f32x4*)srow;
#pragma unroll
      for (int t = 0; t < 32; ++t) {
        f32x4 vv = sv[t];
        vv[0] = __expf(vv[0] - mx); vv[1] = __expf(vv[1] - mx);
        vv[2] = __expf(vv[2] - mx); vv[3] = __expf(vv[3] - mx);
        sum += vv[0] + vv[1] + vv[2] + vv[3];
        svw[t] = vv;
      }
      sum += __shfl_xor(sum, 16, 32);
      float inv = 1.f / sum;
      // pack own half's 4 x 32-col tiles into bf16 A-frags (4xb128/tile)
#pragma unroll
      for (int t = 0; t < 4; ++t) {
        const f32x4* tp = sv + t * 8;
        f32x4 q0 = tp[0] * inv, q1 = tp[1] * inv, q2 = tp[2] * inv, q3 = tp[3] * inv;
        f32x4 q4 = tp[4] * inv, q5 = tp[5] * inv, q6 = tp[6] * inv, q7 = tp[7] * inv;
        u32x4 a0, a1, b0, b1;
        a0[0] = pk2(q0[0], q0[1]); a0[1] = pk2(q0[2], q0[3]);
        a0[2] = pk2(q1[0], q1[1]); a0[3] = pk2(q1[2], q1[3]);   // kk 0..7
        b0[0] = pk2(q2[0], q2[1]); b0[1] = pk2(q2[2], q2[3]);
        b0[2] = pk2(q3[0], q3[1]); b0[3] = pk2(q3[2], q3[3]);   // kk 8..15
        a1[0] = pk2(q4[0], q4[1]); a1[1] = pk2(q4[2], q4[3]);
        a1[2] = pk2(q5[0], q5[1]); a1[3] = pk2(q5[2], q5[3]);   // kk 16..23
        b1[0] = pk2(q6[0], q6[1]); b1[1] = pk2(q6[2], q6[3]);
        b1[2] = pk2(q7[0], q7[1]); b1[3] = pk2(q7[2], q7[3]);   // kk 24..31
        unsigned short* base = afW + (hi * 4 + t) * 512;
        *(u32x4*)(base + row * 16)            = a0;
        *(u32x4*)(base + row * 16 + 8)        = a1;
        *(u32x4*)(base + (row + 16) * 16)     = b0;
        *(u32x4*)(base + (row + 16) * 16 + 8) = b1;
      }
    }
    __syncthreads();

    // ---- o = attn @ v : 4 accumulators per A-frag ----
    {
      const unsigned short* vFh = vF + (size_t)((b * 8 + h) * 8 * 4) * 512;
      v8f o[4]; o[0] = {}; o[1] = {}; o[2] = {}; o[3] = {};
      for (int kt = 0; kt < 8; ++kt) {
        AF a, b0, b1, b2, b3;
        loadFrag(a, afW + kt * 512, lane);
        const unsigned short* vb = vFh + (size_t)kt * 4 * 512;
        loadFrag(b0, vb,        lane);
        loadFrag(b1, vb + 512,  lane);
        loadFrag(b2, vb + 1024, lane);
        loadFrag(b3, vb + 1536, lane);
        o[0] = __builtin_amdgcn_wmma_f32_16x16x32_bf16(false, a.v, false, b0.v, (short)0, o[0], false, false);
        o[1] = __builtin_amdgcn_wmma_f32_16x16x32_bf16(false, a.v, false, b1.v, (short)0, o[1], false, false);
        o[2] = __builtin_amdgcn_wmma_f32_16x16x32_bf16(false, a.v, false, b2.v, (short)0, o[2], false, false);
        o[3] = __builtin_amdgcn_wmma_f32_16x16x32_bf16(false, a.v, false, b3.v, (short)0, o[3], false, false);
      }
      // hie weights for this lane's 8 rows (16 contiguous floats)
      union { f32x4 v[4]; float f[16]; } hb;
      {
        const f32x4* hv = (const f32x4*)(hwW + hi * 16);
        hb.v[0] = hv[0]; hb.v[1] = hv[1]; hb.v[2] = hv[2]; hb.v[3] = hv[3];
      }
#pragma unroll
      for (int nt = 0; nt < 4; ++nt) {
        int d = nt * 16 + nn;
        float bgv = bgV[(size_t)(b * 8 + h) * 64 + d];
        int ii  = h * 64 + d;
        int ktO = ii >> 5, kkO = ii & 31;
#pragma unroll
        for (int r = 0; r < 8; ++r) {
          int mm = r + hi * 8;
          float w0 = hb.f[2 * r], w1 = hb.f[2 * r + 1];
          ocW[ktO * 512 + idxA(mm, kkO)] = f2bf(o[nt][r] * w0 + w1 * bgv);
        }
      }
    }
    __syncthreads();
  }

  // ---- final GEMM: out_tile(16x512) @ Wo(512x640) + bo, groups of 4 tiles ----
  for (int g = 0; g < 10; ++g) {
    v8f c[4]; c[0] = {}; c[1] = {}; c[2] = {}; c[3] = {};
    for (int kt = 0; kt < 16; ++kt) {
      AF a;
      loadFrag(a, ocW + kt * 512, lane);
      const unsigned short* wb = WoF + (size_t)(kt * 40 + g * 4) * 512;
#pragma unroll
      for (int j = 0; j < 4; ++j) {
        AF bb;
        loadFrag(bb, wb + (size_t)j * 512, lane);
        c[j] = __builtin_amdgcn_wmma_f32_16x16x32_bf16(false, a.v, false, bb.v, (short)0, c[j], false, false);
      }
    }
#pragma unroll
    for (int j = 0; j < 4; ++j) {
      int col = (g * 4 + j) * 16 + nn;
      float bias = bo[col];
#pragma unroll
      for (int r = 0; r < 8; ++r) {
        int mm = r + hi * 8;
        out[((size_t)b * 8192 + row0 + mm) * 640 + col] = c[j][r] + bias;
      }
    }
  }
}

// ---------------------------------------------------------------------------
extern "C" void kernel_launch(void* const* d_in, const int* in_sizes, int n_in,
                              void* d_out, int out_size, void* d_ws, size_t ws_size,
                              hipStream_t stream) {
  const float* x    = (const float*)d_in[0];
  const float* ctx  = (const float*)d_in[1];
  const float* Wq   = (const float*)d_in[2];
  const float* Wk   = (const float*)d_in[3];
  const float* Wv   = (const float*)d_in[4];
  const float* Wk_  = (const float*)d_in[5];
  const float* Wv_  = (const float*)d_in[6];
  const float* Wo   = (const float*)d_in[7];
  const float* bo   = (const float*)d_in[8];
  float* out = (float*)d_out;

  // workspace layout (~3.4 MB total)
  char* ws = (char*)d_ws;
  float*          pooled = (float*)         (ws + 0);        //  10240 B
  unsigned short* WqF    = (unsigned short*)(ws + 10240);    // 655360 B
  unsigned short* WoF    = (unsigned short*)(ws + 665600);   // 655360 B
  unsigned short* kFb    = (unsigned short*)(ws + 1320960);  // 1 MiB
  unsigned short* vFb    = (unsigned short*)(ws + 2369536);  // 1 MiB
  float*          hieK   = (float*)         (ws + 3418112);  //  16 KiB
  float*          bgV    = (float*)         (ws + 3434496);  //   8 KiB

  zero_f32<<<10, 256, 0, stream>>>(pooled, 4 * 640);
  pool_sum<<<dim3(64, 4), 640, 0, stream>>>(x, pooled);
  conv_w<<<1280, 256, 0, stream>>>(Wq, 640, 512, 64, WqF);   // 8 heads
  conv_w<<<1280, 256, 0, stream>>>(Wo, 512, 640, 640, WoF);  // 1 "head"
  proj_kv<<<2048, 256, 0, stream>>>(ctx, Wk, Wv, kFb, vFb);
  hie_proj<<<8, 256, 0, stream>>>(ctx, pooled, Wk, Wk_, Wv_, hieK, bgV);
  main_attn<<<dim3(256, 4), 64, SMEM_BYTES, stream>>>(
      x, WqF, kFb, vFb, hieK, bgV, WoF, bo, out);
}
